// CINN_19713899888869
// MI455X (gfx1250) — matrix operand order repaired and checked by reading, not compiled, not run on hardware
//
#include <hip/hip_runtime.h>
#include <hip/hip_bf16.h>
#include <math.h>

typedef __attribute__((ext_vector_type(16))) _Float16 v16h;
typedef __attribute__((ext_vector_type(8)))  _Float16 v8h;
typedef __attribute__((ext_vector_type(8)))  float    v8f;

#define FLOW_L    8
#define FLOW_D    128
#define FLOW_HALF 64
#define FLOW_H    256
#define FLOW_B    32768
#define MROWS     64          // batch rows per workgroup (4 wave32 x 16-row strips)
#define ZSTR      132         // padded f32 stride (528 B/row)
#define HSTR      264         // padded f16 stride (528 B/row) -- same byte stride as ZSTR
#define CLAMP_C      2.0f
#define ATAN_SCALE_C 0.636f

// ---- explicit global address space so B/bias loads lower to global_load_* ----
#define GAS __attribute__((address_space(1)))

__device__ __forceinline__ v8h gload_v8h(const _Float16* p) {
  return *(const GAS v8h*)p;
}
__device__ __forceinline__ float gloadf(const float* p) {
  return *(const GAS float*)p;
}

// ---------------- WMMA fragment loaders (CDNA5 16x16x32 f16 layouts) ----------------

// A (16x32 f16): lane m=L%16 holds row m; K chunk base b=8*(L/16): K=b..b+7 (v0-3), b+16..b+23 (v4-7)
__device__ __forceinline__ v16h load_afrag_f16(const _Float16* base, int stride, int lane, int k0) {
  const int m = lane & 15;
  const int b = k0 + ((lane >> 4) << 3);
  const _Float16* p = base + m * stride + b;
  v8h lo = *(const v8h*)(p);
  v8h hi = *(const v8h*)(p + 16);
  v16h r;
#pragma unroll
  for (int i = 0; i < 8; ++i) { r[i] = lo[i]; r[i + 8] = hi[i]; }
  return r;
}

__device__ __forceinline__ v16h load_afrag_f32(const float* base, int stride, int lane, int k0) {
  const int m = lane & 15;
  const int b = k0 + ((lane >> 4) << 3);
  const float* p = base + m * stride + b;
  float4 x0 = *(const float4*)(p);
  float4 x1 = *(const float4*)(p + 4);
  float4 x2 = *(const float4*)(p + 16);
  float4 x3 = *(const float4*)(p + 20);
  v16h r;
  r[0]  = (_Float16)x0.x; r[1]  = (_Float16)x0.y; r[2]  = (_Float16)x0.z; r[3]  = (_Float16)x0.w;
  r[4]  = (_Float16)x1.x; r[5]  = (_Float16)x1.y; r[6]  = (_Float16)x1.z; r[7]  = (_Float16)x1.w;
  r[8]  = (_Float16)x2.x; r[9]  = (_Float16)x2.y; r[10] = (_Float16)x2.z; r[11] = (_Float16)x2.w;
  r[12] = (_Float16)x3.x; r[13] = (_Float16)x3.y; r[14] = (_Float16)x3.z; r[15] = (_Float16)x3.w;
  return r;
}

// B (32x16 f16): lane holds column c=L%16, K=b..b+7 and b+16..b+23 with b=8*(L/16) (matches A).
// Weights are pre-transposed to [n][K] so this is two contiguous 16B global loads.
__device__ __forceinline__ v16h load_bfrag(const _Float16* Wt, int Kdim, int lane, int n0, int k0) {
  const int c = n0 + (lane & 15);
  const int kb = k0 + ((lane >> 4) << 3);
  const _Float16* p = Wt + c * Kdim + kb;
  v8h lo = gload_v8h(p);
  v8h hi = gload_v8h(p + 16);
  v16h r;
#pragma unroll
  for (int i = 0; i < 8; ++i) { r[i] = lo[i]; r[i + 8] = hi[i]; }
  return r;
}

// ---------------- wave-level GEMM: 16-row strip, bias + ReLU, f16 store to LDS ----------------
// Double-buffered B fragments: loads for tile n+1 issue before the WMMA chain of tile n.
// Entirely wave-local: reads/writes only this wave's 16 rows.

template<int KT, bool AF32>
__device__ __forceinline__ void gemm_store(const void* Asrc, int astride,
                                           const _Float16* __restrict__ Wt,
                                           const float* __restrict__ bias,
                                           _Float16* outlds, int ostride, int N,
                                           int row0, int lane) {
  constexpr int Kdim = KT * 32;
  v16h af[KT];
#pragma unroll
  for (int kt = 0; kt < KT; ++kt) {
    if (AF32)
      af[kt] = load_afrag_f32((const float*)Asrc + row0 * astride, astride, lane, kt * 32);
    else
      af[kt] = load_afrag_f16((const _Float16*)Asrc + row0 * astride, astride, lane, kt * 32);
  }
  const int cl = lane & 15;
  const int rbase = row0 + ((lane >> 4) << 3);
  const int NT = N / 16;   // even (8 or 16)

  v16h b0[KT], b1[KT];
#pragma unroll
  for (int kt = 0; kt < KT; ++kt) b0[kt] = load_bfrag(Wt, Kdim, lane, 0, kt * 32);

  for (int nt = 0; nt < NT; nt += 2) {
    // prefetch tile nt+1 while computing tile nt
#pragma unroll
    for (int kt = 0; kt < KT; ++kt) b1[kt] = load_bfrag(Wt, Kdim, lane, (nt + 1) * 16, kt * 32);
    {
      v8f acc = {};
#pragma unroll
      for (int kt = 0; kt < KT; ++kt)
        acc = __builtin_amdgcn_wmma_f32_16x16x32_f16(false, af[kt], false, b0[kt],
                                                     (short)0, acc, false, false);
      const int col = nt * 16 + cl;
      const float bv = gloadf(bias + col);
#pragma unroll
      for (int v = 0; v < 8; ++v) {
        float t = fmaxf(acc[v] + bv, 0.0f);
        outlds[(rbase + v) * ostride + col] = (_Float16)t;
      }
    }
    // prefetch tile nt+2 while computing tile nt+1
    if (nt + 2 < NT) {
#pragma unroll
      for (int kt = 0; kt < KT; ++kt) b0[kt] = load_bfrag(Wt, Kdim, lane, (nt + 2) * 16, kt * 32);
    }
    {
      v8f acc = {};
#pragma unroll
      for (int kt = 0; kt < KT; ++kt)
        acc = __builtin_amdgcn_wmma_f32_16x16x32_f16(false, af[kt], false, b1[kt],
                                                     (short)0, acc, false, false);
      const int col = (nt + 1) * 16 + cl;
      const float bv = gloadf(bias + col);
#pragma unroll
      for (int v = 0; v < 8; ++v) {
        float t = fmaxf(acc[v] + bv, 0.0f);
        outlds[(rbase + v) * ostride + col] = (_Float16)t;
      }
    }
  }
}

// ---------------- final GEMM + coupling update, straight from accumulators ----------------
// a = h @ Wc^T + bc (64x128); s = 1.272*atan(a[:, :64]); t = a[:, 64:]
// zbuf[:, xcol0:xcol0+64] = exp(s) * zbuf[...] + t ; jacs[row] += sum(s).  Wave-local rows.

template<int KT>
__device__ __forceinline__ void gemm_couple(const _Float16* hlds, int astride,
                                            const _Float16* __restrict__ WcT,
                                            const float* __restrict__ bias,
                                            float* zbuf, int zstride, int xcol0,
                                            float* jacs, int row0, int lane) {
  constexpr int Kdim = KT * 32;
  v16h af[KT];
#pragma unroll
  for (int kt = 0; kt < KT; ++kt)
    af[kt] = load_afrag_f16(hlds + row0 * astride, astride, lane, kt * 32);
  const int cl = lane & 15;
  const int rbase = row0 + ((lane >> 4) << 3);
  float jpart[8];
#pragma unroll
  for (int v = 0; v < 8; ++v) jpart[v] = 0.0f;

  for (int n = 0; n < 4; ++n) {            // s-tile n pairs with t-tile n+4
    // preload all fragments for both output tiles, then run two independent WMMA chains
    v16h bs[KT], bt[KT];
#pragma unroll
    for (int kt = 0; kt < KT; ++kt) {
      bs[kt] = load_bfrag(WcT, Kdim, lane, n * 16, kt * 32);
      bt[kt] = load_bfrag(WcT, Kdim, lane, 64 + n * 16, kt * 32);
    }
    v8f accs = {}, acct = {};
#pragma unroll
    for (int kt = 0; kt < KT; ++kt) {
      accs = __builtin_amdgcn_wmma_f32_16x16x32_f16(false, af[kt], false, bs[kt],
                                                    (short)0, accs, false, false);
      acct = __builtin_amdgcn_wmma_f32_16x16x32_f16(false, af[kt], false, bt[kt],
                                                    (short)0, acct, false, false);
    }
    const int col = n * 16 + cl;
    const float bs_ = gloadf(bias + col);
    const float bt_ = gloadf(bias + 64 + col);
#pragma unroll
    for (int v = 0; v < 8; ++v) {
      const int r = rbase + v;
      float s = CLAMP_C * ATAN_SCALE_C * atanf(accs[v] + bs_);
      float tt = acct[v] + bt_;
      float* zp = &zbuf[r * zstride + xcol0 + col];
      *zp = __expf(s) * (*zp) + tt;
      jpart[v] += s;
    }
  }
#pragma unroll
  for (int v = 0; v < 8; ++v) atomicAdd(&jacs[rbase + v], jpart[v]);   // ds_add_f32 (own rows)
}

// ---------------- weight prep: f32 [ls][K][N] -> f16 transposed [ls][N][K] ----------------

__global__ __launch_bounds__(256) void transpose_cvt(const float* __restrict__ W,
                                                     _Float16* __restrict__ Wt,
                                                     int K, int N, int total) {
  int i = blockIdx.x * 256 + threadIdx.x;
  if (i >= total) return;
  int n = i % N;
  int k = (i / N) % K;
  int ls = i / (N * K);
  Wt[(size_t)ls * K * N + (size_t)n * K + k] = (_Float16)W[i];
}

// ---------------- main flow kernel: 4 fully independent waves, zero barriers ----------------
// All dependencies are wave-local (each wave owns 16 batch rows; the permutation only
// moves columns, never rows), and DS ops within a wave are architecturally in-order.

__global__ __launch_bounds__(128) void flow_kernel(
    const float* __restrict__ x,
    const float* __restrict__ ba, const float* __restrict__ bb,
    const float* __restrict__ bc, const int* __restrict__ perms,
    const _Float16* __restrict__ WaT, const _Float16* __restrict__ WbT,
    const _Float16* __restrict__ WcT,
    float* __restrict__ out) {
  __shared__ __align__(16) float    zbuf[MROWS * ZSTR];   // 33.0 KB  state z (f32)
  __shared__ __align__(16) _Float16 hA[MROWS * HSTR];     // 33.0 KB  hidden (f16)
  __shared__ __align__(16) _Float16 hB[MROWS * HSTR];     // 33.0 KB  hidden (f16)
  __shared__ float jacs[MROWS];

  const int tid  = threadIdx.x;
  const int lane = tid & 31;
  const int wid  = tid >> 5;
  const int row0 = wid * 16;                         // this wave's 16 rows
  const long grow0 = (long)blockIdx.x * MROWS;

  // load this wave's 16 rows of x -> zbuf (float4, 16 iters/lane)
#pragma unroll 4
  for (int j = lane; j < 16 * (FLOW_D / 4); j += 32) {
    const int r = row0 + (j >> 5), cq = j & 31;
    ((float4*)(zbuf + r * ZSTR))[cq] = ((const float4*)(x + (grow0 + r) * FLOW_D))[cq];
  }
  if (lane < 16) jacs[row0 + lane] = 0.0f;

  for (int l = 0; l < FLOW_L; ++l) {
    {   // side 1: a2 = subnet(x2) ; y1 = exp(s2)*x1 + t2   (x2 = z[:,64:128], x1 = z[:,0:64])
      const int ls = l * 2 + 1;
      // warm L2/WGP$ for the other side's big weight panel while we compute this one
      __builtin_prefetch(WbT + (size_t)(l * 2) * (FLOW_H * FLOW_H) + tid * 64, 0, 1);
      gemm_store<2, true>(zbuf + FLOW_HALF, ZSTR, WaT + ls * (FLOW_H * FLOW_HALF),
                          ba + ls * FLOW_H, hA, HSTR, FLOW_H, row0, lane);
      gemm_store<8, false>(hA, HSTR, WbT + ls * (FLOW_H * FLOW_H),
                           bb + ls * FLOW_H, hB, HSTR, FLOW_H, row0, lane);
      gemm_couple<8>(hB, HSTR, WcT + ls * (FLOW_D * FLOW_H),
                     bc + ls * FLOW_D, zbuf, ZSTR, 0, jacs, row0, lane);
    }
    {   // side 0: a1 = subnet(y1) ; y2 = exp(s1)*x2 + t1
      const int ls = l * 2;
      if (l + 1 < FLOW_L)
        __builtin_prefetch(WbT + (size_t)((l + 1) * 2 + 1) * (FLOW_H * FLOW_H) + tid * 64, 0, 1);
      gemm_store<2, true>(zbuf, ZSTR, WaT + ls * (FLOW_H * FLOW_HALF),
                          ba + ls * FLOW_H, hA, HSTR, FLOW_H, row0, lane);
      gemm_store<8, false>(hA, HSTR, WbT + ls * (FLOW_H * FLOW_H),
                           bb + ls * FLOW_H, hB, HSTR, FLOW_H, row0, lane);
      gemm_couple<8>(hB, HSTR, WcT + ls * (FLOW_D * FLOW_H),
                     bc + ls * FLOW_D, zbuf, ZSTR, FLOW_HALF, jacs, row0, lane);
    }
    // permutation: z <- z[:, perm[l]]  (column-only => wave-local rows)
    // f32 bounce view over hA with the SAME 528-byte row stride as the f16 view,
    // so each wave's bounce region coincides with its own hA rows (no cross-wave overlap).
    float* ptmp = (float*)hA;
    const int* pm = perms + l * FLOW_D;
    for (int j = lane; j < 16 * FLOW_D; j += 32) {
      const int r = row0 + (j >> 7), c = j & 127;
      ptmp[r * ZSTR + c] = zbuf[r * ZSTR + c];
    }
    for (int j = lane; j < 16 * FLOW_D; j += 32) {
      const int r = row0 + (j >> 7), c = j & 127;
      zbuf[r * ZSTR + c] = ptmp[r * ZSTR + pm[c]];
    }
  }

  // write this wave's z rows and jac
#pragma unroll 4
  for (int j = lane; j < 16 * (FLOW_D / 4); j += 32) {
    const int r = row0 + (j >> 5), cq = j & 31;
    ((float4*)(out + (grow0 + r) * FLOW_D))[cq] = ((const float4*)(zbuf + r * ZSTR))[cq];
  }
  if (lane < 16) out[(long)FLOW_B * FLOW_D + grow0 + row0 + lane] = jacs[row0 + lane];
}

// ---------------- launch ----------------

extern "C" void kernel_launch(void* const* d_in, const int* in_sizes, int n_in,
                              void* d_out, int out_size, void* d_ws, size_t ws_size,
                              hipStream_t stream) {
  (void)in_sizes; (void)n_in; (void)out_size; (void)ws_size;
  const float* x     = (const float*)d_in[0];
  // d_in[1] = c, d_in[8..11] = cond_* : dead code in reference, skipped.
  const float* Wa    = (const float*)d_in[2];
  const float* ba    = (const float*)d_in[3];
  const float* Wb    = (const float*)d_in[4];
  const float* bb    = (const float*)d_in[5];
  const float* Wc    = (const float*)d_in[6];
  const float* bc    = (const float*)d_in[7];
  const int*   perms = (const int*)d_in[12];
  float* out = (float*)d_out;

  // workspace: f16 transposed weights (total ~3.5 MB, fully L2-resident)
  _Float16* WaT = (_Float16*)d_ws;                       // 16 * 256*64
  _Float16* WbT = WaT + 16 * FLOW_H * FLOW_HALF;         // 16 * 256*256
  _Float16* WcT = WbT + 16 * FLOW_H * FLOW_H;            // 16 * 128*256

  const int totA = FLOW_L * 2 * FLOW_HALF * FLOW_H;      // 262144
  const int totB = FLOW_L * 2 * FLOW_H * FLOW_H;         // 1048576
  const int totC = FLOW_L * 2 * FLOW_H * FLOW_D;         // 524288
  transpose_cvt<<<(totA + 255) / 256, 256, 0, stream>>>(Wa, WaT, FLOW_HALF, FLOW_H, totA);
  transpose_cvt<<<(totB + 255) / 256, 256, 0, stream>>>(Wb, WbT, FLOW_H, FLOW_H, totB);
  transpose_cvt<<<(totC + 255) / 256, 256, 0, stream>>>(Wc, WcT, FLOW_H, FLOW_D, totC);

  flow_kernel<<<FLOW_B / MROWS, 128, 0, stream>>>(x, ba, bb, bc, perms,
                                                  WaT, WbT, WcT, out);
}
